// SelfAttentionV3_10393820857172
// MI455X (gfx1250) — matrix operand order repaired and checked by reading, not compiled, not run on hardware
//
#include <hip/hip_runtime.h>
#include <hip/hip_bf16.h>
#include <stdint.h>

// ---------------------------------------------------------------------------
// CDNA5 (gfx1250) self-attention: X@Wqkv+b -> flash(QK^T softmax @V) -> @Wout+b
// All GEMMs on v_wmma_f32_16x16x32_bf16; double-buffered async global->LDS
// tile pipeline (ASYNCcnt) overlaps copies with WMMA.  LDS is indexed through
// a single addrspace(3) array so fragment reads lower to ds_load_b128 (not
// flat_load) and get partial s_wait_dscnt pipelining.
// ---------------------------------------------------------------------------

typedef __attribute__((ext_vector_type(16))) __bf16 v16bf;
typedef __attribute__((ext_vector_type(8)))  float  v8f;

union Frag16 { v16bf v; uint4 q[2]; };

__device__ __forceinline__ __bf16 f2bf(float f) {
  union { float f; unsigned u; } in; in.f = f;
  unsigned r = in.u + 0x7FFFu + ((in.u >> 16) & 1u);   // round-to-nearest-even
  union { unsigned short s; __bf16 b; } out; out.s = (unsigned short)(r >> 16);
  return out.b;
}

// CDNA5 async copy: 16 bytes global -> LDS per lane, tracked by ASYNCcnt.
__device__ __forceinline__ void async_b128(void* lds_dst, const void* gsrc) {
  asm volatile("global_load_async_to_lds_b128 %0, %1, off"
               :: "v"((uint32_t)(uintptr_t)lds_dst), "v"(gsrc)
               : "memory");
}
__device__ __forceinline__ void wait_async0() {
  asm volatile("s_wait_asynccnt 0" ::: "memory");
}

__device__ __forceinline__ v8f wmma_bf16(const v16bf& a, const v16bf& b, const v8f& c) {
  return __builtin_amdgcn_wmma_f32_16x16x32_bf16(false, a, false, b, (short)0, c,
                                                 false, false);
}

// ---------------------------------------------------------------------------
// fp32 -> bf16 convert (elementwise)
// ---------------------------------------------------------------------------
__global__ void cvt_f32_bf16(const float* __restrict__ in, __bf16* __restrict__ out,
                             long long n) {
  long long i = (long long)blockIdx.x * blockDim.x + threadIdx.x;
  if (i < n) out[i] = f2bf(in[i]);
}

// fp32 [R][C] -> bf16 [C][R]   (weight pre-transpose so GEMM B-frags are contiguous)
__global__ void transpose_cvt(const float* __restrict__ in, __bf16* __restrict__ out,
                              int R, int C) {
  long long i = (long long)blockIdx.x * blockDim.x + threadIdx.x;
  long long total = (long long)R * C;
  if (i < total) {
    int r = (int)(i / C), c = (int)(i % C);
    out[(long long)c * R + r] = f2bf(in[i]);
  }
}

// ---------------------------------------------------------------------------
// GEMM:  C[M][N] = A[M][K](bf16) @ Bt[N][K]^T (bf16) + bias[N]
// Block 256 thr (8 waves) computes 128x128; wave = 32x64 (2x4 wmma tiles).
// Double-buffered async tile pipeline: copy k+1 overlaps compute of k.
// ---------------------------------------------------------------------------
template<bool OUT_F32, bool STORE_VT>
__global__ __launch_bounds__(256) void gemm_bf16(
    const __bf16* __restrict__ A, const __bf16* __restrict__ Bt,
    const float* __restrict__ bias, void* __restrict__ Cout,
    __bf16* __restrict__ Vt, int M, int N, int K)
{
  __shared__ __bf16 sA[2][128 * 32];
  __shared__ __bf16 sB[2][128 * 32];
  const int tid = threadIdx.x;
  const int w = tid >> 5, l = tid & 31, hi = l >> 4, ln = l & 15;
  const int wm = w >> 1, wn = w & 1;
  const int bm = (int)blockIdx.x * 128;
  const int bn = (int)blockIdx.y * 128;

  auto issue_tiles = [&](int buf, int k0) {
    #pragma unroll
    for (int c = 0; c < 2; ++c) {                 // 512 x 16B chunks per tile
      int chunk = tid + c * 256;
      int row = chunk >> 2;
      int co  = (chunk & 3) * 8;
      async_b128(&sA[buf][row * 32 + co], A  + (long long)(bm + row) * K + k0 + co);
      async_b128(&sB[buf][row * 32 + co], Bt + (long long)(bn + row) * K + k0 + co);
    }
  };

  v8f acc[2][4] = {};
  const int NT = K / 32;

  issue_tiles(0, 0);
  for (int kt = 0; kt < NT; ++kt) {
    const int cur = kt & 1;
    wait_async0();            // own tile-k copies done (loads complete in order)
    __syncthreads();          // everyone's copies visible
    if (kt + 1 < NT) issue_tiles(cur ^ 1, (kt + 1) * 32);   // prefetch overlaps

    Frag16 a[2];
    #pragma unroll
    for (int mi = 0; mi < 2; ++mi) {
      int r = wm * 32 + mi * 16 + ln;
      a[mi].q[0] = *(const uint4*)&sA[cur][r * 32 + 8 * hi];
      a[mi].q[1] = *(const uint4*)&sA[cur][r * 32 + 8 * hi + 16];
    }
    #pragma unroll
    for (int ni = 0; ni < 4; ++ni) {
      int nr = wn * 64 + ni * 16 + ln;
      Frag16 b;
      b.q[0] = *(const uint4*)&sB[cur][nr * 32 + 8 * hi];
      b.q[1] = *(const uint4*)&sB[cur][nr * 32 + 8 * hi + 16];
      #pragma unroll
      for (int mi = 0; mi < 2; ++mi)
        acc[mi][ni] = wmma_bf16(a[mi].v, b.v, acc[mi][ni]);
    }
  }

  #pragma unroll
  for (int ni = 0; ni < 4; ++ni) {
    int col = bn + wn * 64 + ni * 16 + ln;
    float bv = bias[col];
    #pragma unroll
    for (int mi = 0; mi < 2; ++mi) {
      int row0 = bm + wm * 32 + mi * 16 + 8 * hi;
      #pragma unroll
      for (int j = 0; j < 8; ++j) {
        float v = acc[mi][ni][j] + bv;
        int row = row0 + j;
        if constexpr (OUT_F32) {
          ((float*)Cout)[(long long)row * N + col] = v;
        } else {
          __bf16 h = f2bf(v);
          ((__bf16*)Cout)[(long long)row * N + col] = h;
          if constexpr (STORE_VT) {
            if (col >= 1536) Vt[(long long)(col - 1536) * M + row] = h;
          }
        }
      }
    }
  }
}

// ---------------------------------------------------------------------------
// Flash attention.  Block = 256 thr (8 waves), 128 queries/block, wave = 16 q.
// K/V streamed in 32-key blocks, double-buffered in 200KB of LDS; Q fragments
// read straight from L2 (global_load_b128).  S=4096, H=768, B=2 hardcoded.
// LDS layout (bf16 elements within one extern-shared array, so ds_* lowering):
//   [0, 24576)           K buf0   32x768
//   [24576, 49152)       K buf1
//   [49152, 73728)       V buf0   768x32 (V transposed)
//   [73728, 98304)       V buf1
//   [98304, 102400)      P staging, 8 waves x (16x32)
// ---------------------------------------------------------------------------
__global__ __launch_bounds__(256) void flash_attn(
    const __bf16* __restrict__ qkv,   // [B*S][2304]  Q|K|V
    const int*    __restrict__ mask,  // [B][S][S]
    const __bf16* __restrict__ Vt,    // [768][B*S]   V transposed
    __bf16*       __restrict__ AO)    // [B*S][768]
{
  extern __shared__ __align__(16) __bf16 smem[];
  constexpr int KSZ = 32 * 768;         // 24576 elements per K/V buffer
  constexpr int VB0 = 2 * KSZ;          // V buffers base
  constexpr int PB0 = 4 * KSZ;          // P staging base

  const int tid = threadIdx.x;
  const int w = tid >> 5, l = tid & 31, hi = l >> 4, ln = l & 15;
  const int batch = (int)blockIdx.x >> 5;
  const int qblk  = (int)blockIdx.x & 31;
  const int tok0  = batch * 4096 + qblk * 128;

  auto issue_kv = [&](int buf, int key0) {
    const int kOff = buf * KSZ;
    const int vOff = VB0 + buf * KSZ;
    #pragma unroll 2
    for (int c = 0; c < 12; ++c) {       // K block: 32 x 768
      int chunk = tid + c * 256;
      int row = chunk / 96;
      int co  = (chunk % 96) * 8;
      async_b128(&smem[kOff + row * 768 + co],
                 qkv + (long long)(batch * 4096 + key0 + row) * 2304 + 768 + co);
    }
    #pragma unroll 2
    for (int c = 0; c < 12; ++c) {       // Vt block: 768 x 32
      int chunk = tid + c * 256;
      int row = chunk >> 2;              // h
      int co  = (chunk & 3) * 8;
      async_b128(&smem[vOff + row * 32 + co],
                 Vt + (long long)row * 8192 + batch * 4096 + key0 + co);
    }
  };

  v8f o[48] = {};
  float mrow[8], lrow[8];
  #pragma unroll
  for (int j = 0; j < 8; ++j) { mrow[j] = -1e30f; lrow[j] = 0.f; }

  const int pOff = PB0 + w * 16 * 32;
  const __bf16* qrow = qkv + (long long)(tok0 + w * 16 + ln) * 2304; // A-frag row
  const float scale = 0.03608439182435161f;   // 1/sqrt(768)

  // warm the cache for this wave's Q row (global_prefetch_b8)
  #pragma unroll 4
  for (int kq = 0; kq < 768; kq += 64) __builtin_prefetch(qrow + kq, 0, 3);

  issue_kv(0, 0);
  for (int kb = 0; kb < 128; ++kb) {
    const int cur  = kb & 1;
    const int key0 = kb * 32;
    const int kOff = cur * KSZ;
    const int vOff = VB0 + cur * KSZ;
    wait_async0();            // own copies for block kb done
    __syncthreads();          // everyone's copies visible
    if (kb + 1 < 128) issue_kv(cur ^ 1, key0 + 32);   // prefetch next block

    // S tile: 16 queries x 32 keys  (Q frags straight from L2)
    v8f s[2] = {};
    for (int kq = 0; kq < 768; kq += 32) {
      Frag16 a;
      a.q[0] = *(const uint4*)(qrow + kq + 8 * hi);
      a.q[1] = *(const uint4*)(qrow + kq + 8 * hi + 16);
      #pragma unroll
      for (int nt = 0; nt < 2; ++nt) {
        int kr = nt * 16 + ln;
        Frag16 b;
        b.q[0] = *(const uint4*)&smem[kOff + kr * 768 + kq + 8 * hi];
        b.q[1] = *(const uint4*)&smem[kOff + kr * 768 + kq + 8 * hi + 16];
        s[nt] = wmma_bf16(a.v, b.v, s[nt]);
      }
    }

    // scale + mask + online softmax update
    float mnew[8], corr8[8];
    #pragma unroll
    for (int j = 0; j < 8; ++j) {
      int qg = qblk * 128 + w * 16 + j + 8 * hi;
      const int* mp = mask + ((long long)batch * 4096 + qg) * 4096;
      float v0 = s[0][j] * scale;
      float v1 = s[1][j] * scale;
      if (mp[key0 + ln] == 0)      v0 = -1e20f;
      if (mp[key0 + 16 + ln] == 0) v1 = -1e20f;
      s[0][j] = v0; s[1][j] = v1;
      float mx = fmaxf(v0, v1);
      #pragma unroll
      for (int d = 1; d < 16; d <<= 1) mx = fmaxf(mx, __shfl_xor(mx, d, 32));
      mnew[j] = fmaxf(mrow[j], mx);
    }
    #pragma unroll
    for (int j = 0; j < 8; ++j) {
      float p0 = __expf(s[0][j] - mnew[j]);
      float p1 = __expf(s[1][j] - mnew[j]);
      float ps = p0 + p1;
      #pragma unroll
      for (int d = 1; d < 16; d <<= 1) ps += __shfl_xor(ps, d, 32);
      float corr = __expf(mrow[j] - mnew[j]);
      lrow[j] = lrow[j] * corr + ps;
      mrow[j] = mnew[j];
      corr8[j] = corr;
      int prow = j + 8 * hi;
      smem[pOff + prow * 32 + ln]      = f2bf(p0);   // C-layout -> A-layout via LDS
      smem[pOff + prow * 32 + 16 + ln] = f2bf(p1);
    }
    #pragma unroll
    for (int nt = 0; nt < 48; ++nt)
      #pragma unroll
      for (int j = 0; j < 8; ++j) o[nt][j] *= corr8[j];

    // O += P (16x32) @ V (32x768)
    Frag16 ap;
    ap.q[0] = *(const uint4*)&smem[pOff + ln * 32 + 8 * hi];
    ap.q[1] = *(const uint4*)&smem[pOff + ln * 32 + 8 * hi + 16];
    #pragma unroll
    for (int nt = 0; nt < 48; ++nt) {
      int h = nt * 16 + ln;
      Frag16 bv;
      bv.q[0] = *(const uint4*)&smem[vOff + h * 32 + 8 * hi];
      bv.q[1] = *(const uint4*)&smem[vOff + h * 32 + 8 * hi + 16];
      o[nt] = wmma_bf16(ap.v, bv.v, o[nt]);
    }
  }

  #pragma unroll
  for (int j = 0; j < 8; ++j) lrow[j] = 1.0f / lrow[j];
  #pragma unroll
  for (int nt = 0; nt < 48; ++nt) {
    int h = nt * 16 + ln;
    #pragma unroll
    for (int j = 0; j < 8; ++j) {
      int tok = tok0 + w * 16 + j + 8 * hi;
      AO[(long long)tok * 768 + h] = f2bf(o[nt][j] * lrow[j]);
    }
  }
}

// ---------------------------------------------------------------------------
// Launch
// ---------------------------------------------------------------------------
extern "C" void kernel_launch(void* const* d_in, const int* in_sizes, int n_in,
                              void* d_out, int out_size, void* d_ws, size_t ws_size,
                              hipStream_t stream) {
  const float* X    = (const float*)d_in[0];   // [2,4096,768]
  const int*   mask = (const int*)  d_in[1];   // [2,4096,4096]
  const float* Wqkv = (const float*)d_in[2];   // [768,2304]
  const float* bqkv = (const float*)d_in[3];   // [2304]
  const float* Wout = (const float*)d_in[4];   // [768,768]
  const float* bout = (const float*)d_in[5];   // [768]
  float* out = (float*)d_out;                  // [2,4096,768] fp32

  const int M = 8192, H = 768, N3 = 2304;

  char* p = (char*)d_ws;
  __bf16* Xb  = (__bf16*)p; p += (long long)M * H * 2;        // 12 MB
  __bf16* Wqt = (__bf16*)p; p += (long long)N3 * H * 2;       // 3.4 MB
  __bf16* Wot = (__bf16*)p; p += (long long)H * H * 2;        // 1.1 MB
  __bf16* QKV = (__bf16*)p; p += (long long)M * N3 * 2;       // 36 MB
  __bf16* Vt  = (__bf16*)p; p += (long long)H * M * 2;        // 12 MB
  __bf16* AO  = (__bf16*)p; p += (long long)M * H * 2;        // 12 MB

  // 1) convert / pre-transpose
  {
    long long n = (long long)M * H;
    cvt_f32_bf16<<<(unsigned)((n + 255) / 256), 256, 0, stream>>>(X, Xb, n);
  }
  {
    long long n = (long long)H * N3;
    transpose_cvt<<<(unsigned)((n + 255) / 256), 256, 0, stream>>>(Wqkv, Wqt, H, N3);
  }
  {
    long long n = (long long)H * H;
    transpose_cvt<<<(unsigned)((n + 255) / 256), 256, 0, stream>>>(Wout, Wot, H, H);
  }

  // 2) QKV = Xb @ Wqkv + b  (also emits V transposed)
  gemm_bf16<false, true><<<dim3(M / 128, N3 / 128), 256, 0, stream>>>(
      Xb, Wqt, bqkv, (void*)QKV, Vt, M, N3, H);

  // 3) flash attention (200 KB dynamic LDS: K/V double-buffered + P staging)
  {
    size_t smem_bytes = (size_t)(4 * 32 * 768 + 8 * 16 * 32) * 2;
    flash_attn<<<64, 256, smem_bytes, stream>>>(QKV, mask, Vt, AO);
  }

  // 4) out = AO @ Wout + b  (fp32 output)
  gemm_bf16<true, false><<<dim3(M / 128, H / 128), 256, 0, stream>>>(
      AO, Wot, bout, (void*)out, nullptr, M, H, H);
}
